// MoEMinGRULayer_35459249996091
// MI455X (gfx1250) — matrix operand order repaired and checked by reading, not compiled, not run on hardware
//
#include <hip/hip_runtime.h>
#include <cstddef>
#include <cstdint>

#define B_    4
#define S_    2048
#define D_    1024
#define M_    4
#define BS    (B_ * S_)
#define CHUNK 64
#define NC    (S_ / CHUNK)

typedef __attribute__((ext_vector_type(2)))  float  v2f;
typedef __attribute__((ext_vector_type(8)))  float  v8f;
typedef __attribute__((ext_vector_type(8)))  __bf16 v8bf;
typedef __attribute__((ext_vector_type(16))) __bf16 v16bf;

// --- bf16 helpers (bit-level, storage-only use of __bf16) ---------------------
__device__ __forceinline__ __bf16 f2bf(float f) {
    union { unsigned short u; __bf16 b; } cv;
    const uint32_t x = __float_as_uint(f);
    cv.u = (unsigned short)((x + 0x7FFFu + ((x >> 16) & 1u)) >> 16);  // RNE
    return cv.b;
}
__device__ __forceinline__ float bf2f(__bf16 h) {
    union { __bf16 b; unsigned short u; } cv;
    cv.b = h;
    return __uint_as_float((uint32_t)cv.u << 16);
}

// ---------------------------------------------------------------------------
// Split x (row-major, [BS, D]) into bf16 hi/lo.
// ---------------------------------------------------------------------------
__global__ void __launch_bounds__(256)
split_x_kernel(const float* __restrict__ x,
               __bf16* __restrict__ hi, __bf16* __restrict__ lo)
{
    const int i = blockIdx.x * blockDim.x + threadIdx.x;
    const float f = x[i];
    const __bf16 h = f2bf(f);
    hi[i] = h;
    lo[i] = f2bf(f - bf2f(h));
}

// ---------------------------------------------------------------------------
// Transpose one stacked weight tensor W[m][k][n] -> Wt[m][n][k] and split into
// bf16 hi/lo. LDS 32x33 tile to keep both sides coalesced.
// ---------------------------------------------------------------------------
__global__ void __launch_bounds__(256)
split_wt_kernel(const float* __restrict__ W,
                __bf16* __restrict__ hi, __bf16* __restrict__ lo)
{
    __shared__ float tile[32][33];
    const int m  = blockIdx.z;
    const int n0 = blockIdx.x * 32;
    const int k0 = blockIdx.y * 32;
    const float* __restrict__ src = W + (size_t)m * D_ * D_;
    #pragma unroll
    for (int j = threadIdx.y; j < 32; j += 8)
        tile[j][threadIdx.x] = src[(size_t)(k0 + j) * D_ + n0 + threadIdx.x];
    __syncthreads();
    const size_t dbase = (size_t)m * D_ * D_;
    #pragma unroll
    for (int j = threadIdx.y; j < 32; j += 8) {
        const float f = tile[threadIdx.x][j];                // = W[k0+tx][n0+j]
        const __bf16 h = f2bf(f);
        const size_t di = dbase + (size_t)(n0 + j) * D_ + k0 + threadIdx.x;
        hi[di] = h;
        lo[di] = f2bf(f - bf2f(h));
    }
}

// ---------------------------------------------------------------------------
// Build a v16bf fragment from two aligned b128 chunks.
// ---------------------------------------------------------------------------
__device__ __forceinline__ v16bf load_frag16(const __bf16* __restrict__ p0,
                                             const __bf16* __restrict__ p1)
{
    const v8bf c1 = *(const v8bf*)p0;
    const v8bf c2 = *(const v8bf*)p1;
    v16bf r;
    #pragma unroll
    for (int i = 0; i < 8; ++i) { r[i] = c1[i]; r[i + 8] = c2[i]; }
    return r;
}

// ---------------------------------------------------------------------------
// Kernel 1: fused 3x GEMM (g,v,d) for all experts via bf16x3 split-precision
// V_WMMA_F32_16X16X32_BF16 (acc = ah*bh + ah*bl + al*bh, fp32 accumulate).
// Each wave: 2 row tiles x 16 cols x 3 matrices (18 WMMAs / K=32 step).
// Writes interleaved (x_scan, a) float2 pairs to workspace [M][BS][D][2].
// ---------------------------------------------------------------------------
__global__ void __launch_bounds__(256)
gemm_act_kernel(const __bf16* __restrict__ xh, const __bf16* __restrict__ xl,
                const __bf16* __restrict__ wgh, const __bf16* __restrict__ wgl,
                const __bf16* __restrict__ wvh, const __bf16* __restrict__ wvl,
                const __bf16* __restrict__ wdh, const __bf16* __restrict__ wdl,
                const float* __restrict__ bg, const float* __restrict__ bv,
                const float* __restrict__ bd, float* __restrict__ xa)
{
    const int wave = threadIdx.x >> 5;
    const int lane = threadIdx.x & 31;
    const int half = lane >> 4;
    const int r16  = lane & 15;

    const int row0 = blockIdx.x * 32;                      // 2 row tiles of 16
    const int col  = (blockIdx.y * 8 + wave) * 16 + r16;   // this lane's N column
    const int m    = blockIdx.z;

    // A: row-major bf16, lane holds K in [8h,8h+8) and [16+8h,16+8h+8)
    const __bf16* __restrict__ ar[2][2];
    ar[0][0] = xh + (size_t)(row0 + r16) * D_;
    ar[0][1] = xl + (size_t)(row0 + r16) * D_;
    ar[1][0] = ar[0][0] + (size_t)16 * D_;
    ar[1][1] = ar[0][1] + (size_t)16 * D_;
    const int aoff0 = 8 * half;
    const int aoff1 = 16 + 8 * half;

    // B: transposed Wt[m][n][k], lane holds K in [8h,8h+8) and [16+8h,16+8h+8) of column n=col
    const size_t wb = (size_t)m * D_ * D_ + (size_t)col * D_;
    const __bf16* __restrict__ wc[3][2] = {
        { wgh + wb, wgl + wb }, { wvh + wb, wvl + wb }, { wdh + wb, wdl + wb } };
    const int boff = 8 * half;

    v8f acc[2][3];
    #pragma unroll
    for (int rt = 0; rt < 2; ++rt)
        #pragma unroll
        for (int mat = 0; mat < 3; ++mat) acc[rt][mat] = (v8f){};

    for (int k0 = 0; k0 < D_; k0 += 32) {
        v16bf a[2][2];
        #pragma unroll
        for (int rt = 0; rt < 2; ++rt)
            #pragma unroll
            for (int p = 0; p < 2; ++p)
                a[rt][p] = load_frag16(ar[rt][p] + k0 + aoff0, ar[rt][p] + k0 + aoff1);

        v16bf b[3][2];
        #pragma unroll
        for (int mat = 0; mat < 3; ++mat)
            #pragma unroll
            for (int p = 0; p < 2; ++p)
                b[mat][p] = load_frag16(wc[mat][p] + k0 + boff,
                                        wc[mat][p] + k0 + boff + 16);

        // terms: (ah,bh), (ah,bl), (al,bh)
        #pragma unroll
        for (int t = 0; t < 3; ++t) {
            const int ai = (t == 2) ? 1 : 0;
            const int bi = (t == 1) ? 1 : 0;
            #pragma unroll
            for (int mat = 0; mat < 3; ++mat)
                #pragma unroll
                for (int rt = 0; rt < 2; ++rt)
                    acc[rt][mat] = __builtin_amdgcn_wmma_f32_16x16x32_bf16(
                        false, a[rt][ai], false, b[mat][bi],
                        (short)0, acc[rt][mat], false, false);
        }
    }

    const float bgs = bg[m * D_ + col];
    const float bvs = bv[m * D_ + col];
    const float bds = bd[m * D_ + col];

    v2f* __restrict__ xap = (v2f*)xa;
    #pragma unroll
    for (int rt = 0; rt < 2; ++rt) {
        #pragma unroll
        for (int r = 0; r < 8; ++r) {
            const int row = row0 + rt * 16 + r + 8 * half;  // C/D: M = r (+8 upper)
            const float g = acc[rt][0][r] + bgs;
            const float v = acc[rt][1][r] + bvs;
            const float d = acc[rt][2][r] + bds;
            const float sg = 1.0f / (1.0f + __expf(-g));
            const float vc = fminf(fmaxf(v, -15.0f), 15.0f);
            const float e2 = __expf(2.0f * vc);
            const float th = (e2 - 1.0f) / (e2 + 1.0f);      // tanh, branchless
            const float sd = 1.0f / (1.0f + __expf(-d));
            v2f pr;
            pr[0] = sg * th;                   // x_scan
            pr[1] = 0.001f + 0.998f * sd;      // a (decay)
            xap[((size_t)m * BS + row) * D_ + col] = pr;
        }
    }
}

// ---------------------------------------------------------------------------
// Kernel 2: router. One wave per token, top-2 + softmax -> dense comb[row][4].
// ---------------------------------------------------------------------------
__global__ void __launch_bounds__(256)
router_kernel(const float* __restrict__ x, const float* __restrict__ gW,
              float* __restrict__ comb)
{
    const int wave = threadIdx.x >> 5;
    const int lane = threadIdx.x & 31;
    const int row  = blockIdx.x * 8 + wave;

    float a0 = 0.f, a1 = 0.f, a2 = 0.f, a3 = 0.f;
    const float* __restrict__ xr = x + (size_t)row * D_;
    for (int d = lane; d < D_; d += 32) {
        const float xv = xr[d];
        const float4 gw = *(const float4*)(gW + (size_t)d * 4);
        a0 += xv * gw.x; a1 += xv * gw.y; a2 += xv * gw.z; a3 += xv * gw.w;
    }
    #pragma unroll
    for (int off = 16; off > 0; off >>= 1) {
        a0 += __shfl_xor(a0, off);
        a1 += __shfl_xor(a1, off);
        a2 += __shfl_xor(a2, off);
        a3 += __shfl_xor(a3, off);
    }
    if (lane == 0) {
        float l[4] = {a0, a1, a2, a3};
        int i0 = 0;
        #pragma unroll
        for (int mm = 1; mm < 4; ++mm) if (l[mm] > l[i0]) i0 = mm;
        int i1 = (i0 == 0) ? 1 : 0;
        #pragma unroll
        for (int mm = 0; mm < 4; ++mm) if (mm != i0 && l[mm] > l[i1]) i1 = mm;
        const float e  = __expf(l[i1] - l[i0]);
        const float w0 = 1.0f / (1.0f + e);
        const float w1 = e * w0;
        float4 c = {0.f, 0.f, 0.f, 0.f};
        ((float*)&c)[i0] = w0;
        ((float*)&c)[i1] = w1;
        *(float4*)(comb + (size_t)row * 4) = c;
    }
}

// ---------------------------------------------------------------------------
// Kernel 3: per-chunk local scan -> carries (prod(a), h_end).
// ---------------------------------------------------------------------------
__global__ void __launch_bounds__(256)
scan_chunk_kernel(const float* __restrict__ xa,
                  float* __restrict__ carryA, float* __restrict__ carryH)
{
    const int d     = blockIdx.x * blockDim.x + threadIdx.x;
    const int chunk = blockIdx.y;
    const int mb    = blockIdx.z;                 // m*B + b
    const int m = mb >> 2, b = mb & 3;

    const v2f* __restrict__ p =
        (const v2f*)xa + ((size_t)m * BS + (size_t)b * S_ + (size_t)chunk * CHUNK) * D_ + d;

    float pa = 1.0f, h = 0.0f;
    for (int s = 0; s < CHUNK; ++s) {
        const v2f t = p[(size_t)s * D_];
        h  = t[1] * h + t[0];
        pa *= t[1];
    }
    const size_t ci = ((size_t)mb * NC + chunk) * D_ + d;
    carryA[ci] = pa;
    carryH[ci] = h;
}

// ---------------------------------------------------------------------------
// Kernel 4: cross-chunk carry scan -> h at the start of each chunk.
// ---------------------------------------------------------------------------
__global__ void __launch_bounds__(256)
scan_carry_kernel(const float* __restrict__ carryA, const float* __restrict__ carryH,
                  float* __restrict__ prefixH)
{
    const int idx = blockIdx.x * blockDim.x + threadIdx.x;  // 0 .. M*B*D-1
    const int d   = idx & (D_ - 1);
    const int mb  = idx >> 10;
    float h = 0.0f;
    for (int c = 0; c < NC; ++c) {
        const size_t ci = ((size_t)mb * NC + c) * D_ + d;
        prefixH[ci] = h;
        h = carryA[ci] * h + carryH[ci];
    }
}

// ---------------------------------------------------------------------------
// Kernel 5: final re-scan seeded with chunk prefixes + top-k combine.
// ---------------------------------------------------------------------------
__global__ void __launch_bounds__(256)
scan_final_kernel(const float* __restrict__ xa, const float* __restrict__ comb,
                  const float* __restrict__ prefixH, float* __restrict__ out)
{
    const int d     = blockIdx.x * blockDim.x + threadIdx.x;
    const int chunk = blockIdx.y;
    const int b     = blockIdx.z;

    float h[M_];
    #pragma unroll
    for (int m = 0; m < M_; ++m)
        h[m] = prefixH[((size_t)(m * B_ + b) * NC + chunk) * D_ + d];

    const int row0 = b * S_ + chunk * CHUNK;
    for (int s = 0; s < CHUNK; ++s) {
        const int row = row0 + s;
        const float4 c = *(const float4*)(comb + (size_t)row * 4);
        float acc = 0.0f;
        #pragma unroll
        for (int m = 0; m < M_; ++m) {
            const v2f t = *((const v2f*)xa + ((size_t)m * BS + row) * D_ + d);
            h[m] = t[1] * h[m] + t[0];
            acc += ((const float*)&c)[m] * h[m];
        }
        out[(size_t)row * D_ + d] = acc;
    }
}

// ---------------------------------------------------------------------------
extern "C" void kernel_launch(void* const* d_in, const int* in_sizes, int n_in,
                              void* d_out, int out_size, void* d_ws, size_t ws_size,
                              hipStream_t stream)
{
    (void)in_sizes; (void)n_in; (void)out_size; (void)ws_size;

    const float* x  = (const float*)d_in[0];
    const float* Wg = (const float*)d_in[1];
    const float* bg = (const float*)d_in[2];
    const float* Wv = (const float*)d_in[3];
    const float* bv = (const float*)d_in[4];
    const float* Wd = (const float*)d_in[5];
    const float* bd = (const float*)d_in[6];
    const float* gW = (const float*)d_in[7];
    float* out = (float*)d_out;

    // Workspace carve-up (byte based)
    char* w = (char*)d_ws;
    float* xa      = (float*)w; w += (size_t)2 * M_ * BS * D_ * sizeof(float);
    float* comb    = (float*)w; w += (size_t)BS * 4 * sizeof(float);
    float* carryA  = (float*)w; w += (size_t)M_ * B_ * NC * D_ * sizeof(float);
    float* carryH  = (float*)w; w += (size_t)M_ * B_ * NC * D_ * sizeof(float);
    float* prefixH = (float*)w; w += (size_t)M_ * B_ * NC * D_ * sizeof(float);
    __bf16* xhi = (__bf16*)w; w += (size_t)BS * D_ * 2;
    __bf16* xlo = (__bf16*)w; w += (size_t)BS * D_ * 2;
    __bf16* wgh = (__bf16*)w; w += (size_t)M_ * D_ * D_ * 2;
    __bf16* wgl = (__bf16*)w; w += (size_t)M_ * D_ * D_ * 2;
    __bf16* wvh = (__bf16*)w; w += (size_t)M_ * D_ * D_ * 2;
    __bf16* wvl = (__bf16*)w; w += (size_t)M_ * D_ * D_ * 2;
    __bf16* wdh = (__bf16*)w; w += (size_t)M_ * D_ * D_ * 2;
    __bf16* wdl = (__bf16*)w; w += (size_t)M_ * D_ * D_ * 2;

    // 0) Split/transpose pre-passes (one-time, L2/HBM trivial)
    split_x_kernel<<<(BS * D_) / 256, 256, 0, stream>>>(x, xhi, xlo);
    dim3 tb(32, 8);
    dim3 tg(D_ / 32, D_ / 32, M_);
    split_wt_kernel<<<tg, tb, 0, stream>>>(Wg, wgh, wgl);
    split_wt_kernel<<<tg, tb, 0, stream>>>(Wv, wvh, wvl);
    split_wt_kernel<<<tg, tb, 0, stream>>>(Wd, wdh, wdl);

    // 1) Fused expert GEMMs + activations (bf16x3 WMMA, fp32 accumulate)
    dim3 g1(BS / 32, D_ / 128, M_);
    gemm_act_kernel<<<g1, 256, 0, stream>>>(xhi, xlo, wgh, wgl, wvh, wvl,
                                            wdh, wdl, bg, bv, bd, xa);

    // 2) Router
    router_kernel<<<BS / 8, 256, 0, stream>>>(x, gW, comb);

    // 3) Chunked linear scan
    dim3 g3(D_ / 256, NC, M_ * B_);
    scan_chunk_kernel<<<g3, 256, 0, stream>>>(xa, carryA, carryH);
    scan_carry_kernel<<<(M_ * B_ * D_) / 256, 256, 0, stream>>>(carryA, carryH, prefixH);

    // 4) Final re-scan + expert combine
    dim3 g5(D_ / 256, NC, B_);
    scan_final_kernel<<<g5, 256, 0, stream>>>(xa, comb, prefixH, out);
}
